// SemiARDecoder_62577673503615
// MI455X (gfx1250) — matrix operand order repaired and checked by reading, not compiled, not run on hardware
//
#include <hip/hip_runtime.h>
#include <hip/hip_bf16.h>

typedef __attribute__((ext_vector_type(16))) _Float16 v16h;
typedef __attribute__((ext_vector_type(8)))  _Float16 v8h;
typedef __attribute__((ext_vector_type(8)))  float    v8f;

#define NTOT  131072
#define HDIM  256
#define DDIM  32
#define NOUT  4
#define NITER 3
#define MROWS 32   // samples per wave (2 WMMA M-tiles)

// packed weight offsets in d_ws, in _Float16 units (each tile = 512 halves = 1KB)
#define OFF_WIH 0        // 48 ntiles x 3 ktiles
#define OFF_WHH 73728    // 48 x 8
#define OFF_WP1 270336   // 16 x 19
#define OFF_WP2 425984   // 16 x 8
#define OFF_WP3 491520   // 1  x 8
#define OFF_WF1 495616   // 16 x 19
#define OFF_WF2 651264   // 16 x 8
#define OFF_WF3 716800   // 1  x 8

// per-workgroup (1 wave) LDS layout, bytes
#define OFF_HT   0        // 32x256 f16 h_t_int
#define OFF_HG   16384    // 32x256 f16 recurrent h
#define OFF_H1   32768    // 32x256 f16 scratch (h_new / h1)
#define OFF_XB   49152    // 32x96  f16 [dt|te|pad]
#define OFF_H2S  55296    // 32x32  f16 h2 staging
#define OFF_ELAP 57344    // 32 f32
#define OFF_DTM  57472    // 32 f32
#define OFF_PSUM 57600    // 32x4 f32
#define SHMEM_BYTES 58112

// global (device memory) address-space qualified pointer types, so packed-B
// traffic lowers to global_load_b128 (LOADcnt only) instead of flat_load
// (which ties up the LDS/DScnt path we need for A-tile ds_load_b128).
typedef const __attribute__((address_space(1))) _Float16* gph;
typedef const __attribute__((address_space(1))) v16h*     gpv16;

// ---------------- device helpers ----------------

__device__ __forceinline__ v8f wmma_f16(v16h a, v16h b, v8f c) {
  return __builtin_amdgcn_wmma_f32_16x16x32_f16(false, a, false, b, (short)0, c,
                                                false, false);
}

__device__ __forceinline__ float sigf(float x) {
  return 1.0f / (1.0f + expf(-x));
}

// A-matrix tile (16x32 f16) from row-major LDS buffer, per ISA A layout:
// lanes 0-15: K = kbase+0..7 and kbase+16..23 ; lanes 16-31: +8
__device__ __forceinline__ v16h load_a(const _Float16* buf, int stride, int kbase,
                                       int lane) {
  int m = lane & 15;
  int g = (lane >> 4) << 3; // 0 or 8
  const _Float16* p = buf + m * stride + kbase + g;
  v8h lo = *(const v8h*)(p);
  v8h hi = *(const v8h*)(p + 16);
  return __builtin_shufflevector(lo, hi, 0, 1, 2, 3, 4, 5, 6, 7,
                                 8, 9, 10, 11, 12, 13, 14, 15);
}

// B-matrix tile (32x16 f16) pre-packed so each lane reads 32 contiguous bytes
__device__ __forceinline__ v16h load_b(const _Float16* w, int lane) {
  gph g = (gph)w;
  return *(gpv16)(g + lane * 16);
}

// Fill xb rows: col0 = dt, cols 1..32 = mixed sin, cols 33..64 = mixed cos.
// Cols 65..95 stay zero (padding, initialized once).
__device__ __forceinline__ void fill_te(_Float16* xb, const float* dtv,
                                        const float* TE_w, const float* sw,
                                        const float* sb, float lam, int s0,
                                        int lane) {
  xb[lane * 96] = (_Float16)dtv[lane];
  for (int idx = lane; idx < MROWS * 32; idx += 32) {
    int row = idx >> 5, d = idx & 31;
    float dt = dtv[row];
    float ind = dt * TE_w[(size_t)(s0 + row) * DDIM + d];
    float sh = dt * sw[d] + sb[d];
    float sn = (1.0f - lam) * sinf(ind) + lam * sinf(sh);
    float cs = (1.0f - lam) * cosf(ind) + lam * cosf(sh);
    xb[row * 96 + 1 + d] = (_Float16)sn;
    xb[row * 96 + 33 + d] = (_Float16)cs;
  }
}

// One GRU step for 32 samples: x (65 cols, padded 96) in xb, h in hG (32x256).
// Writes new h to h1b then copies back to hG. Biases folded into the tail.
__device__ __forceinline__ void gru_step(const _Float16* xb, _Float16* hG,
                                         _Float16* h1b, const _Float16* wih,
                                         const _Float16* whh, const float* b_ih,
                                         const float* b_hh, int lane) {
  int nn = lane & 15;
  int hi = lane >> 4;
  for (int t = 0; t < 16; ++t) {
    int n = t * 16 + nn;
    v8f air0 = {}, aiz0 = {}, ain0 = {}, ahr0 = {}, ahz0 = {}, ahn0 = {};
    v8f air1 = {}, aiz1 = {}, ain1 = {}, ahr1 = {}, ahz1 = {}, ahn1 = {};
#pragma unroll
    for (int kt = 0; kt < 3; ++kt) {
      v16h a0 = load_a(xb, 96, kt * 32, lane);
      v16h a1 = load_a(xb + 16 * 96, 96, kt * 32, lane);
      v16h br = load_b(wih + ((size_t)(t)*3 + kt) * 512, lane);
      v16h bz = load_b(wih + ((size_t)(t + 16) * 3 + kt) * 512, lane);
      v16h bn = load_b(wih + ((size_t)(t + 32) * 3 + kt) * 512, lane);
      air0 = wmma_f16(a0, br, air0); air1 = wmma_f16(a1, br, air1);
      aiz0 = wmma_f16(a0, bz, aiz0); aiz1 = wmma_f16(a1, bz, aiz1);
      ain0 = wmma_f16(a0, bn, ain0); ain1 = wmma_f16(a1, bn, ain1);
    }
#pragma unroll
    for (int kt = 0; kt < 8; ++kt) {
      v16h a0 = load_a(hG, HDIM, kt * 32, lane);
      v16h a1 = load_a(hG + 16 * HDIM, HDIM, kt * 32, lane);
      v16h br = load_b(whh + ((size_t)(t)*8 + kt) * 512, lane);
      v16h bz = load_b(whh + ((size_t)(t + 16) * 8 + kt) * 512, lane);
      v16h bn = load_b(whh + ((size_t)(t + 32) * 8 + kt) * 512, lane);
      ahr0 = wmma_f16(a0, br, ahr0); ahr1 = wmma_f16(a1, br, ahr1);
      ahz0 = wmma_f16(a0, bz, ahz0); ahz1 = wmma_f16(a1, bz, ahz1);
      ahn0 = wmma_f16(a0, bn, ahn0); ahn1 = wmma_f16(a1, bn, ahn1);
    }
    float brz = b_ih[n] + b_hh[n];
    float bzz = b_ih[HDIM + n] + b_hh[HDIM + n];
    float bin = b_ih[2 * HDIM + n];
    float bhn = b_hh[2 * HDIM + n];
#pragma unroll
    for (int r = 0; r < 8; ++r) {
      int m = r + 8 * hi;
      {
        float rg = sigf(air0[r] + ahr0[r] + brz);
        float z = sigf(aiz0[r] + ahz0[r] + bzz);
        float nv = tanhf(ain0[r] + bin + rg * (ahn0[r] + bhn));
        float ho = (float)hG[m * HDIM + n];
        h1b[m * HDIM + n] = (_Float16)((1.0f - z) * nv + z * ho);
      }
      {
        int m2 = m + 16;
        float rg = sigf(air1[r] + ahr1[r] + brz);
        float z = sigf(aiz1[r] + ahz1[r] + bzz);
        float nv = tanhf(ain1[r] + bin + rg * (ahn1[r] + bhn));
        float ho = (float)hG[m2 * HDIM + n];
        h1b[m2 * HDIM + n] = (_Float16)((1.0f - z) * nv + z * ho);
      }
    }
  }
  for (int i = lane; i < (MROWS * HDIM) / 2; i += 32)
    ((unsigned int*)hG)[i] = ((const unsigned int*)h1b)[i];
}

// MLP3 over inp = [dt|te (xb, K 0..95) , h_t_int (hT, K 96..351) , ht (hG, K 352..607)]
__device__ __forceinline__ void mlp3_step(
    const _Float16* xb, const _Float16* hT, const _Float16* hG, _Float16* h1b,
    _Float16* h2s, float* psum, const _Float16* W1, const float* b1,
    const _Float16* W2, const float* b2, const _Float16* W3, const float* b3,
    bool sig, float* outp, int s0, int k, int lane) {
  int nn = lane & 15, hi = lane >> 4;
  // layer 1: 16 output tiles x 19 K tiles, both M-halves per B load
  for (int nt = 0; nt < 16; ++nt) {
    int n = nt * 16 + nn;
    v8f acc0 = {}, acc1 = {};
#pragma unroll
    for (int kt = 0; kt < 19; ++kt) {
      const _Float16* src; int stride, kb;
      if (kt < 3)       { src = xb; stride = 96;   kb = kt * 32; }
      else if (kt < 11) { src = hT; stride = HDIM; kb = (kt - 3) * 32; }
      else              { src = hG; stride = HDIM; kb = (kt - 11) * 32; }
      v16h a0 = load_a(src, stride, kb, lane);
      v16h a1 = load_a(src + 16 * stride, stride, kb, lane);
      v16h b = load_b(W1 + ((size_t)nt * 19 + kt) * 512, lane);
      acc0 = wmma_f16(a0, b, acc0);
      acc1 = wmma_f16(a1, b, acc1);
    }
    float b1v = b1[n];
#pragma unroll
    for (int r = 0; r < 8; ++r) {
      int m = r + 8 * hi;
      h1b[m * HDIM + n] = (_Float16)fmaxf(acc0[r] + b1v, 0.0f);
      h1b[(m + 16) * HDIM + n] = (_Float16)fmaxf(acc1[r] + b1v, 0.0f);
    }
  }
  // layer 2 fused into layer 3 via 32x32 staging tile
  v8f acc3_0 = {}, acc3_1 = {};
  for (int q = 0; q < 8; ++q) {
#pragma unroll
    for (int half = 0; half < 2; ++half) {
      int nt = q * 2 + half;
      int n = nt * 16 + nn;
      v8f acc0 = {}, acc1 = {};
#pragma unroll
      for (int kt = 0; kt < 8; ++kt) {
        v16h a0 = load_a(h1b, HDIM, kt * 32, lane);
        v16h a1 = load_a(h1b + 16 * HDIM, HDIM, kt * 32, lane);
        v16h b = load_b(W2 + ((size_t)nt * 8 + kt) * 512, lane);
        acc0 = wmma_f16(a0, b, acc0);
        acc1 = wmma_f16(a1, b, acc1);
      }
      float b2v = b2[n];
#pragma unroll
      for (int r = 0; r < 8; ++r) {
        int m = r + 8 * hi;
        h2s[m * 32 + half * 16 + nn] = (_Float16)fmaxf(acc0[r] + b2v, 0.0f);
        h2s[(m + 16) * 32 + half * 16 + nn] = (_Float16)fmaxf(acc1[r] + b2v, 0.0f);
      }
    }
    v16h a0 = load_a(h2s, 32, 0, lane);
    v16h a1 = load_a(h2s + 16 * 32, 32, 0, lane);
    v16h b = load_b(W3 + (size_t)q * 512, lane);
    acc3_0 = wmma_f16(a0, b, acc3_0);
    acc3_1 = wmma_f16(a1, b, acc3_1);
  }
  // write outputs (cols 0..3 valid)
  if (nn < NOUT) {
    float b3v = b3[nn];
#pragma unroll
    for (int r = 0; r < 8; ++r) {
      int m = r + 8 * hi;
      float v0 = acc3_0[r] + b3v;
      float v1 = acc3_1[r] + b3v;
      if (sig) {
        v0 = 1.0f / (1.0f + expf(-v0));
        v1 = 1.0f / (1.0f + expf(-v1));
      }
      outp[(size_t)(s0 + m) * 12 + k * NOUT + nn] = v0;
      outp[(size_t)(s0 + m + 16) * 12 + k * NOUT + nn] = v1;
      if (sig) {
        psum[m * 4 + nn] = v0;
        psum[(m + 16) * 4 + nn] = v1;
      }
    }
  }
}

// ---------------- kernels ----------------

// Pack f32 weight [nrows x kcols] into f16 B-tile layout:
// dst[tile][lane][j] = W[nt*16 + lane%16][kt*32 + 8*(lane/16) + (j%8) + 16*(j/8)]
// kmode 0: zero-pad K >= kcols. kmode 1 (Wp1/Wf1): K<65->K, 65..95->zero, >=96->K-31
__global__ void pack_b_kernel(const float* __restrict__ src,
                              _Float16* __restrict__ dst, int ntiles, int ktiles,
                              int nrows, int kcols, int kmode) {
  int total = ntiles * ktiles * 512;
  for (int i = blockIdx.x * blockDim.x + threadIdx.x; i < total;
       i += gridDim.x * blockDim.x) {
    int tile = i >> 9;
    int rem = i & 511;
    int lane = rem >> 4;
    int j = rem & 15;
    int kt = tile % ktiles;
    int nt = tile / ktiles;
    int n = nt * 16 + (lane & 15);
    int kk = kt * 32 + 8 * (lane >> 4) + (j & 7) + 16 * (j >> 3);
    int ks;
    if (kmode == 1)
      ks = (kk < 65) ? kk : ((kk < 96) ? -1 : kk - 31);
    else
      ks = (kk < kcols) ? kk : -1;
    float v = 0.0f;
    if (n < nrows && ks >= 0) v = src[(size_t)n * kcols + ks];
    dst[i] = (_Float16)v;
  }
}

__global__ __launch_bounds__(32) void semiar_main(
    const float* __restrict__ h_t_int, const float* __restrict__ dt_init,
    const float* __restrict__ Y, const float* __restrict__ TE_w,
    const float* __restrict__ shared_w, const float* __restrict__ shared_b,
    const float* __restrict__ TE_lam, const float* __restrict__ b_ih,
    const float* __restrict__ b_hh, const float* __restrict__ bp1,
    const float* __restrict__ bp2, const float* __restrict__ bp3,
    const float* __restrict__ bf1, const float* __restrict__ bf2,
    const float* __restrict__ bf3, const _Float16* __restrict__ wpk,
    float* __restrict__ outp) {
  extern __shared__ __align__(16) char smem[];
  int lane = threadIdx.x & 31;
  int s0 = blockIdx.x * MROWS;

  _Float16* hT = (_Float16*)(smem + OFF_HT);
  _Float16* hG = (_Float16*)(smem + OFF_HG);
  _Float16* h1b = (_Float16*)(smem + OFF_H1);
  _Float16* xb = (_Float16*)(smem + OFF_XB);
  _Float16* h2s = (_Float16*)(smem + OFF_H2S);
  float* elap = (float*)(smem + OFF_ELAP);
  float* dtm = (float*)(smem + OFF_DTM);
  float* psum = (float*)(smem + OFF_PSUM);

  float tl = TE_lam[0];
  float lam = expf(-tl * tl);

  for (int i = lane; i < MROWS * HDIM; i += 32) {
    int r = i >> 8, c = i & 255;
    _Float16 v = (_Float16)h_t_int[(size_t)(s0 + r) * HDIM + c];
    hT[i] = v;
    hG[i] = v;
  }
  for (int i = lane; i < MROWS * 96; i += 32) xb[i] = (_Float16)0.0f;
  {
    float d0 = dt_init[s0 + lane];
    elap[lane] = d0;
    dtm[lane] = d0;
  }

  const _Float16* wih = wpk + OFF_WIH;
  const _Float16* whh = wpk + OFF_WHH;

  // ---- PERIOD branch ----
  for (int k = 0; k < NITER; ++k) {
    fill_te(xb, elap, TE_w, shared_w, shared_b, lam, s0, lane);
    gru_step(xb, hG, h1b, wih, whh, b_ih, b_hh, lane);
    fill_te(xb, dtm, TE_w, shared_w, shared_b, lam, s0, lane);
    mlp3_step(xb, hT, hG, h1b, h2s, psum, wpk + OFF_WP1, bp1, wpk + OFF_WP2,
              bp2, wpk + OFF_WP3, bp3, true, outp, s0, k, lane);
    {
      float st = psum[lane * 4 + 0] + psum[lane * 4 + 1] + psum[lane * 4 + 2] +
                 psum[lane * 4 + 3];
      elap[lane] = st;
      dtm[lane] += st;
    }
  }

  // ---- FLOW branch ----
  for (int i = lane; i < MROWS * HDIM; i += 32) hG[i] = hT[i];
  elap[lane] = dt_init[s0 + lane];
  for (int k = 0; k < NITER; ++k) {
    fill_te(xb, elap, TE_w, shared_w, shared_b, lam, s0, lane);
    gru_step(xb, hG, h1b, wih, whh, b_ih, b_hh, lane);
    dtm[lane] = Y[(size_t)(s0 + lane) * 36 + (k * NOUT) * 3 + 2];
    fill_te(xb, dtm, TE_w, shared_w, shared_b, lam, s0, lane);
    mlp3_step(xb, hT, hG, h1b, h2s, psum, wpk + OFF_WF1, bf1, wpk + OFF_WF2,
              bf2, wpk + OFF_WF3, bf3, false, outp + (size_t)NTOT * 12, s0, k,
              lane);
    {
      float s = 0.0f;
      for (int j = 0; j < NOUT; ++j)
        s += Y[(size_t)(s0 + lane) * 36 + (k * NOUT + j) * 3 + 0];
      elap[lane] = s;
    }
  }
}

// ---------------- launch ----------------

extern "C" void kernel_launch(void* const* d_in, const int* in_sizes, int n_in,
                              void* d_out, int out_size, void* d_ws,
                              size_t ws_size, hipStream_t stream) {
  const float* h_t_int = (const float*)d_in[0];
  const float* dt_init = (const float*)d_in[1];
  const float* Y = (const float*)d_in[2];
  const float* TE_w = (const float*)d_in[3];
  const float* shared_w = (const float*)d_in[4];
  const float* shared_b = (const float*)d_in[5];
  const float* TE_lam = (const float*)d_in[6];
  const float* W_ih = (const float*)d_in[7];
  const float* b_ih = (const float*)d_in[8];
  const float* W_hh = (const float*)d_in[9];
  const float* b_hh = (const float*)d_in[10];
  const float* Wp1 = (const float*)d_in[11];
  const float* bp1 = (const float*)d_in[12];
  const float* Wp2 = (const float*)d_in[13];
  const float* bp2 = (const float*)d_in[14];
  const float* Wp3 = (const float*)d_in[15];
  const float* bp3 = (const float*)d_in[16];
  const float* Wf1 = (const float*)d_in[17];
  const float* bf1 = (const float*)d_in[18];
  const float* Wf2 = (const float*)d_in[19];
  const float* bf2 = (const float*)d_in[20];
  const float* Wf3 = (const float*)d_in[21];
  const float* bf3 = (const float*)d_in[22];

  _Float16* wpk = (_Float16*)d_ws;
  float* outp = (float*)d_out;

  auto pack = [&](const float* src, size_t off, int nt, int kt, int nr, int kc,
                  int kmode) {
    int total = nt * kt * 512;
    pack_b_kernel<<<(total + 255) / 256, 256, 0, stream>>>(src, wpk + off, nt,
                                                           kt, nr, kc, kmode);
  };
  pack(W_ih, OFF_WIH, 48, 3, 768, 65, 0);
  pack(W_hh, OFF_WHH, 48, 8, 768, 256, 0);
  pack(Wp1, OFF_WP1, 16, 19, 256, 577, 1);
  pack(Wp2, OFF_WP2, 16, 8, 256, 256, 0);
  pack(Wp3, OFF_WP3, 1, 8, 4, 256, 0);
  pack(Wf1, OFF_WF1, 16, 19, 256, 577, 1);
  pack(Wf2, OFF_WF2, 16, 8, 256, 256, 0);
  pack(Wf3, OFF_WF3, 1, 8, 4, 256, 0);

  int blocks = NTOT / MROWS; // 4096
  semiar_main<<<blocks, 32, SHMEM_BYTES, stream>>>(
      h_t_int, dt_init, Y, TE_w, shared_w, shared_b, TE_lam, b_ih, b_hh, bp1,
      bp2, bp3, bf1, bf2, bf3, wpk, outp);
}
